// SNN_MLP_56770877718717
// MI455X (gfx1250) — compile-verified, matrix operand order checked by reading, not run on hardware
//
#include <hip/hip_runtime.h>

typedef __attribute__((ext_vector_type(16))) _Float16 v16h;
typedef __attribute__((ext_vector_type(8)))  _Float16 v8h;
typedef __attribute__((ext_vector_type(8)))  float    v8f;

#define TSTEPS 2048
#define XS  2056          // padded input row stride (floats): 3 + 2048 + pad
#define S1S 136           // s1 row stride (halfs)  -> 272B rows, conflict-free b128
#define S2S 264           // s2 row stride (halfs)  -> 528B rows, conflict-free b128
#define PARTS 272         // per-wave partial tile (16*17 floats)

// LDS carve (bytes)
#define OFF_XS   0
#define OFF_S1   131584   // 16*2056*4
#define OFF_S2   135936   // +16*136*2
#define OFF_PART 144384   // +16*264*2
#define OFF_V3   153088   // +8*272*4
#define OFF_ACC  153728   // +160*4
#define SMEM_BYTES 154368 // +160*4

__launch_bounds__(256, 1)
__global__ void snn_scan_kernel(const float* __restrict__ x,       // [64,2048]
                                const float* __restrict__ conv_w,  // [128,1,7]
                                const float* __restrict__ W1,      // [256,128]
                                const float* __restrict__ W2,      // [10,256]
                                float* __restrict__ out)           // [64,10]
{
    extern __shared__ char smem[];
    float*    xs   = (float*)(smem + OFF_XS);
    _Float16* s1   = (_Float16*)(smem + OFF_S1);
    _Float16* s2   = (_Float16*)(smem + OFF_S2);
    float*    part = (float*)(smem + OFF_PART);
    float*    v3s  = (float*)(smem + OFF_V3);
    float*    accs = (float*)(smem + OFF_ACC);

    const int tid  = threadIdx.x;
    const int wave = tid >> 5;
    const int lane = tid & 31;
    const int nlo  = lane & 15;     // WMMA column/row-within-tile index
    const int g    = lane >> 4;     // lane group (0/1)
    const int r0   = blockIdx.x * 16;

    // ---- stage this block's 16 input rows into LDS, zero-padded by 3 each side
    for (int i = tid; i < 16 * 2048; i += 256) {
        int b = i >> 11, t = i & 2047;
        xs[b * XS + 3 + t] = x[(r0 + b) * 2048 + t];
    }
    if (tid < 16) {
        xs[tid*XS + 0] = 0.f; xs[tid*XS + 1] = 0.f; xs[tid*XS + 2] = 0.f;
        xs[tid*XS + 2051] = 0.f; xs[tid*XS + 2052] = 0.f; xs[tid*XS + 2053] = 0.f;
        xs[tid*XS + 2054] = 0.f; xs[tid*XS + 2055] = 0.f;
    }

    // ---- per-thread conv state: row brow, 8 channels [c8, c8+8)
    const int brow = tid >> 4;
    const int c8   = (tid & 15) * 8;
    float cw[8][7];
    #pragma unroll
    for (int j = 0; j < 8; ++j)
        #pragma unroll
        for (int kk = 0; kk < 7; ++kk)
            cw[j][kk] = conv_w[(c8 + j) * 7 + kk];

    // ---- resident W1 B-fragments (f16). wave w owns N-tiles {2w, 2w+1}.
    // B[k,n] = W1[n,k]; 16x16x32 16-bit fragment: lane (n=l%16,g=l/16),
    // halves 0..7 -> k = kt*32 + g*8 + j ; halves 8..15 -> k = kt*32 + 16 + g*8 + j
    v16h b1[2][4];
    #pragma unroll
    for (int nt2 = 0; nt2 < 2; ++nt2) {
        int n = (wave * 2 + nt2) * 16 + nlo;
        #pragma unroll
        for (int kt = 0; kt < 4; ++kt) {
            v16h f;
            #pragma unroll
            for (int j = 0; j < 8; ++j) {
                f[j]     = (_Float16)W1[n * 128 + kt * 32 + g * 8 + j];
                f[8 + j] = (_Float16)W1[n * 128 + kt * 32 + 16 + g * 8 + j];
            }
            b1[nt2][kt] = f;
        }
    }

    // ---- resident W2 B-fragment: wave == K-tile; columns n>=10 zero-padded
    v16h b2;
    #pragma unroll
    for (int j = 0; j < 8; ++j) {
        int klo = wave * 32 + g * 8 + j;
        int khi = wave * 32 + 16 + g * 8 + j;
        b2[j]     = (nlo < 10) ? (_Float16)W2[nlo * 256 + klo] : (_Float16)0.0f;
        b2[8 + j] = (nlo < 10) ? (_Float16)W2[nlo * 256 + khi] : (_Float16)0.0f;
    }

    // ---- persistent LIF state
    float v1r[8];
    #pragma unroll
    for (int j = 0; j < 8; ++j) v1r[j] = 0.f;
    v8f v2r[2] = {};
    if (tid < 160) { v3s[tid] = 0.f; accs[tid] = 0.f; }

    // sliding conv window: xw will hold xs[brow][t .. t+6] after the in-loop shift
    float xw[7];
    #pragma unroll
    for (int k = 1; k < 7; ++k) xw[k] = 0.f;   // overwritten below after sync
    __syncthreads();
    #pragma unroll
    for (int k = 1; k < 7; ++k) xw[k] = xs[brow * XS + (k - 1)];

    for (int t = 0; t < TSTEPS; ++t) {
        // ---------------- phase 1: conv + LIF1, emit s1 (f16) ----------------
        #pragma unroll
        for (int k = 0; k < 6; ++k) xw[k] = xw[k + 1];
        xw[6] = xs[brow * XS + t + 6];

        v8h sv;
        #pragma unroll
        for (int j = 0; j < 8; ++j) {
            float a = 0.f;
            #pragma unroll
            for (int kk = 0; kk < 7; ++kk) a = fmaf(cw[j][kk], xw[kk], a);
            float v = fmaf(v1r[j], 0.5f, a);       // v = v*DECAY + x
            float s = (v >= 1.0f) ? 1.0f : 0.0f;   // fire
            v1r[j] = v - v * s;                    // hard reset
            sv[j] = (_Float16)s;
        }
        *(v8h*)&s1[brow * S1S + c8] = sv;          // one ds_store_b128
        __syncthreads();

        // ---------------- phase 2: GEMM1 (WMMA) + LIF2, emit s2 --------------
        v16h a1[4];
        #pragma unroll
        for (int kt = 0; kt < 4; ++kt) {
            v8h lo = *(const v8h*)&s1[nlo * S1S + kt * 32 + g * 8];
            v8h hi = *(const v8h*)&s1[nlo * S1S + kt * 32 + 16 + g * 8];
            a1[kt] = __builtin_shufflevector(lo, hi,
                     0,1,2,3,4,5,6,7,8,9,10,11,12,13,14,15);
        }
        #pragma unroll
        for (int nt2 = 0; nt2 < 2; ++nt2) {
            v8f c = {};
            #pragma unroll
            for (int kt = 0; kt < 4; ++kt)
                c = __builtin_amdgcn_wmma_f32_16x16x32_f16(
                        false, a1[kt], false, b1[nt2][kt], (short)0, c, false, false);
            #pragma unroll
            for (int i = 0; i < 8; ++i) {
                float v = fmaf(v2r[nt2][i], 0.5f, c[i]);
                float s = (v >= 1.0f) ? 1.0f : 0.0f;
                v2r[nt2][i] = v - v * s;
                int mm = i + 8 * g;
                s2[mm * S2S + (wave * 2 + nt2) * 16 + nlo] = (_Float16)s;
            }
        }
        __syncthreads();

        // ---------------- phase 3: GEMM2 K-split across waves ----------------
        {
            v8h lo = *(const v8h*)&s2[nlo * S2S + wave * 32 + g * 8];
            v8h hi = *(const v8h*)&s2[nlo * S2S + wave * 32 + 16 + g * 8];
            v16h a2 = __builtin_shufflevector(lo, hi,
                      0,1,2,3,4,5,6,7,8,9,10,11,12,13,14,15);
            v8f c2 = {};
            c2 = __builtin_amdgcn_wmma_f32_16x16x32_f16(
                     false, a2, false, b2, (short)0, c2, false, false);
            #pragma unroll
            for (int i = 0; i < 8; ++i)
                part[wave * PARTS + (i + 8 * g) * 17 + nlo] = c2[i];
        }
        __syncthreads();

        // ---------------- phase 4: reduce partials + LIF3 + accumulate -------
        if (tid < 160) {
            int mm = tid / 10, nn = tid % 10;
            float y = 0.f;
            #pragma unroll
            for (int w = 0; w < 8; ++w) y += part[w * PARTS + mm * 17 + nn];
            float v = fmaf(v3s[tid], 0.5f, y);
            float s = (v >= 1.0f) ? 1.0f : 0.0f;
            v3s[tid] = v - v * s;
            accs[tid] += s;
        }
        __syncthreads();
    }

    if (tid < 160)
        out[(r0 + tid / 10) * 10 + (tid % 10)] = accs[tid] * (1.0f / (float)TSTEPS);
}

extern "C" void kernel_launch(void* const* d_in, const int* in_sizes, int n_in,
                              void* d_out, int out_size, void* d_ws, size_t ws_size,
                              hipStream_t stream) {
    (void)in_sizes; (void)n_in; (void)out_size; (void)d_ws; (void)ws_size;
    const float* x      = (const float*)d_in[0];
    const float* conv_w = (const float*)d_in[1];
    const float* W1     = (const float*)d_in[2];
    const float* W2     = (const float*)d_in[3];
    float* out = (float*)d_out;

    // >64KB dynamic LDS: raise the cap (idempotent, deterministic, host-side)
    hipFuncSetAttribute((const void*)snn_scan_kernel,
                        hipFuncAttributeMaxDynamicSharedMemorySize, SMEM_BYTES);

    snn_scan_kernel<<<4, 256, SMEM_BYTES, stream>>>(x, conv_w, W1, W2, out);
}
